// Transformer_13451837571760
// MI455X (gfx1250) — compile-verified, matrix operand order checked by reading, not run on hardware
//
#include <hip/hip_runtime.h>

// ---------------------------------------------------------------------------
// CDNA5 / gfx1250 transformer forward. wave32, WMMA bf16 -> f32 accum.
// One wave owns one 16x16 output tile. Whole working set (~100MB) fits in the
// 192MB L2, so fragment gathers go straight to global (L2-resident) with no
// LDS staging. All WMMA branches are wave-uniform (EXEC all-ones).
// ---------------------------------------------------------------------------

typedef __attribute__((ext_vector_type(16))) __bf16 v16bf;
typedef __attribute__((ext_vector_type(8)))  float  v8f;

#define NEGINF (-1e9f)
#define INV_SQRT32 0.17677669529663687f

// A-fragment (16x32, 16-bit): lane l<16 -> row l, K {0..7,16..23};
// lane l>=16 -> row l-16, K {8..15,24..31}.  (ISA 7.12.2)
__device__ __forceinline__ v16bf load_a_frag(const float* __restrict__ X, int ld,
                                             int row0, int k0, int lane) {
    int half = lane >> 4, r = lane & 15;
    const float* p = X + (size_t)(row0 + r) * ld + k0 + half * 8;
    v16bf a;
#pragma unroll
    for (int i = 0; i < 8; ++i) a[i] = (__bf16)p[i];
#pragma unroll
    for (int i = 0; i < 8; ++i) a[8 + i] = (__bf16)p[16 + i];
    return a;
}

// B-fragment (32x16, 16-bit), W row-major [K x N]: lane l<16 -> col l,
// K {0..7,16..23}; lane l>=16 -> col l-16, K {8..15,24..31}.
__device__ __forceinline__ v16bf load_b_frag(const float* __restrict__ W, int ld,
                                             int k0, int col0, int lane) {
    int half = lane >> 4, c = lane & 15;
    const float* p = W + (size_t)(k0 + half * 8) * ld + col0 + c;
    v16bf b;
#pragma unroll
    for (int i = 0; i < 8; ++i) b[i] = (__bf16)p[(size_t)i * ld];
#pragma unroll
    for (int i = 0; i < 8; ++i) b[8 + i] = (__bf16)p[(size_t)(16 + i) * ld];
    return b;
}

// C/D layout (ISA 7.12.2): VGPR r, lane<16 -> (M=r, N=lane); lane>=16 -> (M=r+8).
__device__ __forceinline__ void store_tile(float* __restrict__ Y, int ld, int row0,
                                           int col0, int lane, v8f acc,
                                           const float* __restrict__ bias,
                                           int relu, float scale) {
    int half = lane >> 4, c = lane & 15;
    int n = col0 + c;
    float bv = bias ? bias[n] : 0.0f;
#pragma unroll
    for (int r = 0; r < 8; ++r) {
        int m = row0 + r + half * 8;
        float v = acc[r] + bv;
        if (relu) v = fmaxf(v, 0.0f);
        Y[(size_t)m * ld + n] = v * scale;
    }
}

// Y[M,N] = act(X[M,K] @ W[K,N] + b) * scale, all fp32 in memory.
__global__ void gemm_bias_act(const float* __restrict__ X, const float* __restrict__ W,
                              const float* __restrict__ bias, float* __restrict__ Y,
                              int M, int N, int K, int relu, float scale) {
    int lane = threadIdx.x & 31;
    int gw = (int)((blockIdx.x * blockDim.x + threadIdx.x) >> 5);
    int tilesN = N >> 4, tilesM = M >> 4;
    if (gw >= tilesM * tilesN) return;               // wave-uniform
    int tm = gw / tilesN, tn = gw % tilesN;
    int row0 = tm * 16, col0 = tn * 16;
    v8f acc = {0.f, 0.f, 0.f, 0.f, 0.f, 0.f, 0.f, 0.f};
    for (int k0 = 0; k0 < K; k0 += 32) {
        v16bf a = load_a_frag(X, K, row0, k0, lane);
        v16bf b = load_b_frag(W, N, k0, col0, lane);
        acc = __builtin_amdgcn_wmma_f32_16x16x32_bf16(false, a, false, b,
                                                      (short)0, acc, false, false);
    }
    store_tile(Y, N, row0, col0, lane, acc, bias, relu, scale);
}

// logits[b,h,i,j] = (K[b,i,h*32:]·Q[b,j,h*32:]) / sqrt(32) + mask[b,j]*NEG,
// optionally then l*w_at + w_hb*tab[b,i,j] + w_pi*tab_pi[b,i,j].
// d_head == 32 -> exactly one WMMA per 16x16 tile.
__global__ void attn_logits(const float* __restrict__ Kp, const float* __restrict__ Qp,
                            const int* __restrict__ mask, float* __restrict__ out,
                            int Lk, int Lq,
                            const float* __restrict__ tab, const float* __restrict__ tabpi,
                            const float* __restrict__ aug_at, const float* __restrict__ aug_hb,
                            const float* __restrict__ aug_pi, int layer) {
    int lane = threadIdx.x & 31;
    int gw = (int)((blockIdx.x * blockDim.x + threadIdx.x) >> 5);
    int tI = Lk >> 4, tJ = Lq >> 4;
    int per = tI * tJ;
    if (gw >= 16 * per) return;                      // B*H = 16, wave-uniform
    int bh = gw / per, r2 = gw % per;
    int ti = r2 / tJ, tj = r2 % tJ;
    int b = bh >> 3, h = bh & 7;
    int i0 = ti * 16, j0 = tj * 16;

    v16bf a  = load_a_frag(Kp + (size_t)b * Lk * 256, 256, i0, h * 32, lane);
    // Q^T fragment: B[k=d][n=j] = Q[j, h*32+d] -> mirrored layout == A-load on Q rows.
    v16bf bt = load_a_frag(Qp + (size_t)b * Lq * 256, 256, j0, h * 32, lane);
    v8f acc = {0.f, 0.f, 0.f, 0.f, 0.f, 0.f, 0.f, 0.f};
    acc = __builtin_amdgcn_wmma_f32_16x16x32_bf16(false, a, false, bt,
                                                  (short)0, acc, false, false);

    int half = lane >> 4, c = lane & 15;
    int j = j0 + c;
    float mt = NEGINF * (float)mask[(size_t)b * Lq + j];
    bool aug = (tab != nullptr);
    float wat = 1.f, whb = 0.f, wpi = 0.f;
    if (aug) { wat = aug_at[layer]; whb = aug_hb[layer]; wpi = aug_pi[layer]; }
    float* obase = out + (size_t)bh * Lk * Lq;
#pragma unroll
    for (int r = 0; r < 8; ++r) {
        int i = i0 + r + half * 8;
        float l = acc[r] * INV_SQRT32 + mt;
        if (aug)
            l = l * wat + whb * tab[((size_t)b * Lk + i) * Lq + j]
                        + wpi * tabpi[((size_t)b * Lk + i) * Lq + j];
        obase[(size_t)i * Lq + j] = l;
    }
}

// Joint softmax over the whole Lk*Lq plane per (b,h). One block per (b,h).
__global__ void joint_softmax(float* __restrict__ p, int n) {
    __shared__ float red[1024];
    float* base = p + (size_t)blockIdx.x * n;
    int t = threadIdx.x, nt = blockDim.x;
    float m = -3.4e38f;
    for (int i = t; i < n; i += nt) m = fmaxf(m, base[i]);
    red[t] = m; __syncthreads();
    for (int s = 512; s > 0; s >>= 1) { if (t < s) red[t] = fmaxf(red[t], red[t + s]); __syncthreads(); }
    m = red[0]; __syncthreads();
    float sum = 0.f;
    for (int i = t; i < n; i += nt) { float e = expf(base[i] - m); base[i] = e; sum += e; }
    red[t] = sum; __syncthreads();
    for (int s = 512; s > 0; s >>= 1) { if (t < s) red[t] += red[t + s]; __syncthreads(); }
    float inv = 1.0f / red[0]; __syncthreads();
    for (int i = t; i < n; i += nt) base[i] *= inv;
}

// Yc[b, i, h*32+d] = sum_j P[b,h,i,j] * V[b, j, h*32+d]   (head-concat layout)
__global__ void attn_av(const float* __restrict__ P, const float* __restrict__ V,
                        float* __restrict__ Yc, int Li, int Lj) {
    int lane = threadIdx.x & 31;
    int gw = (int)((blockIdx.x * blockDim.x + threadIdx.x) >> 5);
    int tI = Li >> 4;
    int per = tI * 2;                                // d_head=32 -> 2 col tiles
    if (gw >= 16 * per) return;
    int bh = gw / per, r2 = gw % per;
    int ti = r2 >> 1, td = r2 & 1;
    int b = bh >> 3, h = bh & 7;
    int i0 = ti * 16, d0 = td * 16;
    const float* Pb = P + (size_t)bh * Li * Lj;
    const float* Vb = V + (size_t)b * Lj * 256;
    v8f acc = {0.f, 0.f, 0.f, 0.f, 0.f, 0.f, 0.f, 0.f};
    for (int j0 = 0; j0 < Lj; j0 += 32) {
        v16bf a = load_a_frag(Pb, Lj, i0, j0, lane);
        v16bf bf = load_b_frag(Vb, 256, j0, h * 32 + d0, lane);
        acc = __builtin_amdgcn_wmma_f32_16x16x32_bf16(false, a, false, bf,
                                                      (short)0, acc, false, false);
    }
    store_tile(Yc + (size_t)b * Li * 256, 256, i0, h * 32 + d0, lane, acc, nullptr, 0, 1.0f);
}

// y = LN(x + a) * g + beta, row width 256. One block per row.
__global__ void add_layernorm(const float* __restrict__ x, const float* __restrict__ a,
                              const float* __restrict__ g, const float* __restrict__ beta,
                              float* __restrict__ y, float eps) {
    __shared__ float red[256];
    int row = blockIdx.x, t = threadIdx.x;
    size_t idx = (size_t)row * 256 + t;
    float v = x[idx] + a[idx];
    red[t] = v; __syncthreads();
    for (int s = 128; s > 0; s >>= 1) { if (t < s) red[t] += red[t + s]; __syncthreads(); }
    float mu = red[0] * (1.0f / 256.0f); __syncthreads();
    float d = v - mu;
    red[t] = d * d; __syncthreads();
    for (int s = 128; s > 0; s >>= 1) { if (t < s) red[t] += red[t + s]; __syncthreads(); }
    float var = red[0] * (1.0f / 256.0f);
    y[idx] = d * rsqrtf(var + eps) * g[t] + beta[t];
}

// final = softmax(y[:,0,:] @ Wf + bf), written to out[0..3]. Tiny scalar kernel.
__global__ void final_head(const float* __restrict__ y, const float* __restrict__ W,
                           const float* __restrict__ bias, float* __restrict__ out, int Lpro) {
    int b = threadIdx.x;
    if (b < 2) {
        const float* row = y + (size_t)b * Lpro * 256;
        float l0 = bias[0], l1 = bias[1];
        for (int d = 0; d < 256; ++d) { float v = row[d]; l0 += v * W[d * 2 + 0]; l1 += v * W[d * 2 + 1]; }
        float m = fmaxf(l0, l1);
        float e0 = expf(l0 - m), e1 = expf(l1 - m);
        float s = e0 + e1;
        out[b * 2 + 0] = e0 / s;
        out[b * 2 + 1] = e1 / s;
    }
}

// ---------------------------------------------------------------------------
// Input flattening assumption: setup_inputs() dict insertion order, params
// flattened depth-first in insertion order (lists by index, {W,b}, {q,k,v,o},
// {g,b}).
//   0 rna[2,1024,640] 1 pvecs[2,512,1024] 2 m_rna(i32) 3 m_pro(i32)
//   4 m_cross(i32) 5 stat_pot[2,1,512,1024] 6 stat_pot_pi
//   7 enc_first.W 8 enc_first.b 9 dec_first.W 10 dec_first.b
//   enc layer l @ 11+16l: qW qb kW kb vW vb oW ob f1W f1b f2W f2b ln1g ln1b ln2g ln2b
//   dec layer l @ 43+26l: mha1(8) mha2(8) f1W f1b f2W f2b ln1g ln1b ln2g ln2b ln3g ln3b
//   95 aug_hb[2] 96 aug_pi[2] 97 aug_at[2] 98 final.W 99 final.b
// ---------------------------------------------------------------------------
extern "C" void kernel_launch(void* const* d_in, const int* in_sizes, int n_in,
                              void* d_out, int out_size, void* d_ws, size_t ws_size,
                              hipStream_t stream) {
    (void)in_sizes; (void)n_in; (void)out_size; (void)ws_size;
    const float* rna    = (const float*)d_in[0];
    const float* pvec   = (const float*)d_in[1];
    const int*   mrna   = (const int*)d_in[2];
    const int*   mpro   = (const int*)d_in[3];
    const int*   mcross = (const int*)d_in[4];
    const float* tab    = (const float*)d_in[5];
    const float* tabpi  = (const float*)d_in[6];
    auto PF = [&](int i) { return (const float*)d_in[i]; };

    // workspace carve-up (floats)
    float* w = (float*)d_ws;
    float* encx = w;  w += (size_t)2048 * 256;   // encoder x / final enc output
    float* qb   = w;  w += (size_t)2048 * 256;
    float* kb   = w;  w += (size_t)2048 * 256;
    float* vb   = w;  w += (size_t)2048 * 256;
    float* cat  = w;  w += (size_t)2048 * 256;   // attn head-concat output
    float* t1   = w;  w += (size_t)2048 * 256;
    float* t2   = w;  w += (size_t)2048 * 256;
    float* ffn  = w;  w += (size_t)2048 * 1024;
    float* decy = w;  w += (size_t)1024 * 256;
    float* probs = w; w += (size_t)16 * 1024 * 1024;  // self-attn scratch
    float* out = (float*)d_out;

    auto gemm = [&](const float* X, int pw, int pb, float* Y, int M, int N, int K,
                    int relu, float scale) {
        int waves = (M >> 4) * (N >> 4);
        gemm_bias_act<<<(waves + 7) / 8, 256, 0, stream>>>(X, PF(pw), PF(pb), Y,
                                                           M, N, K, relu, scale);
    };
    auto lnadd = [&](const float* x, const float* a, int pg, int pb, float* y,
                     int rows, float eps) {
        add_layernorm<<<rows, 256, 0, stream>>>(x, a, PF(pg), PF(pb), y, eps);
    };

    const float stem_scale = 0.0625f;  // sqrt(256)/256

    // ---------------- encoder ----------------
    gemm(rna, 7, 8, encx, 2048, 256, 640, 1, stem_scale);
    for (int l = 0; l < 2; ++l) {
        int pb = 11 + l * 16;
        gemm(encx, pb + 0, pb + 1, qb, 2048, 256, 256, 0, 1.f);
        gemm(encx, pb + 2, pb + 3, kb, 2048, 256, 256, 0, 1.f);
        gemm(encx, pb + 4, pb + 5, vb, 2048, 256, 256, 0, 1.f);
        { int waves = 16 * 64 * 64;
          attn_logits<<<(waves + 7) / 8, 256, 0, stream>>>(kb, qb, mrna, probs, 1024, 1024,
              nullptr, nullptr, nullptr, nullptr, nullptr, 0); }
        joint_softmax<<<16, 1024, 0, stream>>>(probs, 1024 * 1024);
        { int waves = 16 * 64 * 2;
          attn_av<<<(waves + 7) / 8, 256, 0, stream>>>(probs, vb, cat, 1024, 1024); }
        gemm(cat, pb + 6, pb + 7, t1, 2048, 256, 256, 0, 1.f);
        lnadd(encx, t1, pb + 12, pb + 13, t2, 2048, 1e-6f);           // o1
        gemm(t2, pb + 8, pb + 9, ffn, 2048, 1024, 256, 1, 1.f);
        gemm(ffn, pb + 10, pb + 11, t1, 2048, 256, 1024, 0, 1.f);
        lnadd(t2, t1, pb + 14, pb + 15, encx, 2048, 1e-6f);           // next x
    }

    // ---------------- decoder ----------------
    gemm(pvec, 9, 10, decy, 1024, 256, 1024, 1, stem_scale);
    for (int l = 0; l < 2; ++l) {
        int pb = 43 + l * 26;
        // self-attention (mha1)
        gemm(decy, pb + 0, pb + 1, qb, 1024, 256, 256, 0, 1.f);
        gemm(decy, pb + 2, pb + 3, kb, 1024, 256, 256, 0, 1.f);
        gemm(decy, pb + 4, pb + 5, vb, 1024, 256, 256, 0, 1.f);
        { int waves = 16 * 32 * 32;
          attn_logits<<<(waves + 7) / 8, 256, 0, stream>>>(kb, qb, mpro, probs, 512, 512,
              nullptr, nullptr, nullptr, nullptr, nullptr, 0); }
        joint_softmax<<<16, 1024, 0, stream>>>(probs, 512 * 512);
        { int waves = 16 * 32 * 2;
          attn_av<<<(waves + 7) / 8, 256, 0, stream>>>(probs, vb, cat, 512, 512); }
        gemm(cat, pb + 6, pb + 7, t1, 1024, 256, 256, 0, 1.f);
        lnadd(t1, decy, pb + 20, pb + 21, t2, 1024, 1e-3f);           // o1
        // cross-attention (mha2): v=enc, q=enc, k=o1 (reference's arg order)
        gemm(encx, pb + 8,  pb + 9,  qb, 2048, 256, 256, 0, 1.f);
        gemm(t2,   pb + 10, pb + 11, kb, 1024, 256, 256, 0, 1.f);
        gemm(encx, pb + 12, pb + 13, vb, 2048, 256, 256, 0, 1.f);
        float* aw = out + 4 + (size_t)l * 2 * 8 * 512 * 1024;         // aws[l]
        { int waves = 16 * 32 * 64;
          attn_logits<<<(waves + 7) / 8, 256, 0, stream>>>(kb, qb, mcross, aw, 512, 1024,
              tab, tabpi, PF(97), PF(95), PF(96), l); }
        joint_softmax<<<16, 1024, 0, stream>>>(aw, 512 * 1024);
        { int waves = 16 * 32 * 2;
          attn_av<<<(waves + 7) / 8, 256, 0, stream>>>(aw, vb, cat, 512, 1024); }
        gemm(cat, pb + 14, pb + 15, t1, 1024, 256, 256, 0, 1.f);
        lnadd(t1, t2, pb + 22, pb + 23, t2, 1024, 1e-3f);             // o2 (in place)
        gemm(t2, pb + 16, pb + 17, ffn, 1024, 1024, 256, 1, 1.f);
        gemm(ffn, pb + 18, pb + 19, t1, 1024, 256, 1024, 0, 1.f);
        lnadd(t1, t2, pb + 24, pb + 25, decy, 1024, 1e-3f);           // next y
    }

    final_head<<<1, 32, 0, stream>>>(decy, PF(98), PF(99), out, 512);
}
